// MultiHeadAttention_22342419873888
// MI455X (gfx1250) — compile-verified, matrix operand order checked by reading, not run on hardware
//
#include <hip/hip_runtime.h>
#include <hip/hip_bf16.h>

// Problem constants (B,T,E,H,D) = (4,2048,512,8,64)
#define TDIM 2048
#define EDIM 512
#define HDIM 8
#define DDIM 64
#define BDIM 4
#define BH   (BDIM * HDIM)   // 32
#define MROWS (BDIM * TDIM)  // 8192

typedef __bf16 bf16x16 __attribute__((ext_vector_type(16)));
typedef float  f32x8   __attribute__((ext_vector_type(8)));

union FragBF {
    uint4   u[2];
    bf16x16 v;
};

__device__ __forceinline__ unsigned short f2bf(float f) {
    unsigned int u = __float_as_uint(f);
    u += 0x7FFFu + ((u >> 16) & 1u);     // round-to-nearest-even
    return (unsigned short)(u >> 16);
}

__device__ __forceinline__ f32x8 wmma_bf16(const FragBF& a, const FragBF& b, f32x8 c) {
    return __builtin_amdgcn_wmma_f32_16x16x32_bf16(
        /*neg_a=*/false, a.v, /*neg_b=*/false, b.v,
        /*c_mod=*/(short)0, c, /*reuse_a=*/false, /*reuse_b=*/false);
}

// ---------------------------------------------------------------------------
// Async global->LDS copy (CDNA5 GLOBAL_LOAD_ASYNC_TO_LDS_B128, ASYNCcnt),
// with a synchronous load+ds_store fallback if the builtin is unavailable.
// Builtin signature (from hipcc diagnostic): first param is
// 'int __vector(4) addrspace(1)*' (global, non-const); LDS dst is addrspace(3).
// ---------------------------------------------------------------------------
#if defined(__AMDGCN__)
#if __has_builtin(__builtin_amdgcn_global_load_async_to_lds_b128) && \
    __has_builtin(__builtin_amdgcn_s_wait_asynccnt)
#define USE_ASYNC_LDS 1
#endif
#endif

typedef int v4i __attribute__((vector_size(16)));
typedef __attribute__((address_space(1))) v4i* gas_v4i_ptr;
typedef __attribute__((address_space(3))) v4i* las_v4i_ptr;

__device__ __forceinline__ void cp16_async(unsigned short* l, const unsigned short* g) {
#ifdef USE_ASYNC_LDS
    void* gv = (void*)g;                 // strip const
    void* lv = (void*)l;
    __builtin_amdgcn_global_load_async_to_lds_b128(
        (gas_v4i_ptr)gv, (las_v4i_ptr)lv, 0, 0);
#else
    *(uint4*)l = *(const uint4*)g;
#endif
}

template <int N>
__device__ __forceinline__ void wait_async() {
#ifdef USE_ASYNC_LDS
    __builtin_amdgcn_s_wait_asynccnt(N);
#endif
}

// ---------------------------------------------------------------------------
// Kernel 1: fp32 -> bf16 conversion of x (B*T, E)
// ---------------------------------------------------------------------------
__global__ __launch_bounds__(256) void k_cvt_x(const float* __restrict__ x,
                                               unsigned short* __restrict__ xb, int n) {
    int i = (blockIdx.x * 256 + threadIdx.x) * 2;
    if (i < n) {
        float2 f = *(const float2*)(x + i);
        unsigned int packed = (unsigned int)f2bf(f.x) | ((unsigned int)f2bf(f.y) << 16);
        *(unsigned int*)(xb + i) = packed;
    }
}

// ---------------------------------------------------------------------------
// Kernel 2: transpose+convert weights (in,out)fp32 -> (out,in)bf16 so the
// WMMA B-fragment (per-lane N, contiguous K) is a contiguous 16B load.
// ---------------------------------------------------------------------------
__global__ __launch_bounds__(256) void k_cvt_w(const float* __restrict__ Wq,
                                               const float* __restrict__ Wk,
                                               const float* __restrict__ Wv,
                                               const float* __restrict__ Wo,
                                               unsigned short* __restrict__ wt) {
    int i = blockIdx.x * 256 + threadIdx.x;   // 0 .. 512*512-1, i = n*512 + k
    int mat = blockIdx.y;
    const float* src = (mat == 0) ? Wq : (mat == 1) ? Wk : (mat == 2) ? Wv : Wo;
    int n = i >> 9, k = i & 511;
    wt[mat * EDIM * EDIM + i] = f2bf(src[k * EDIM + n]);
}

// ---------------------------------------------------------------------------
// Kernel 3: QKV projection GEMM (8192x512 @ 512x512) + bias, bf16 WMMA.
// The 64x32 weight tile is shared by all 8 waves -> stage it in LDS once per
// k-step via async copy, double buffered. A-frags (per-wave rows) from global.
// z=0 -> Q (B,H,T,D); z=1 -> K (B,H,T,D); z=2 -> V transposed (B,H,D,T).
// ---------------------------------------------------------------------------
__global__ __launch_bounds__(256) void k_qkv(const unsigned short* __restrict__ xb,
                                             const unsigned short* __restrict__ wt,
                                             const float* __restrict__ bq,
                                             const float* __restrict__ bk,
                                             const float* __restrict__ bv,
                                             unsigned short* __restrict__ Qb,
                                             unsigned short* __restrict__ Kb,
                                             unsigned short* __restrict__ Vt) {
    __shared__ unsigned short wtile[2][64 * 40];   // 64 n-rows x 32 k, padded
    const int tid  = threadIdx.x;
    const int lane = tid & 31;
    const int wave = tid >> 5;
    const int z    = blockIdx.z;
    const int m0   = blockIdx.x * 128 + wave * 16;
    const int n0   = blockIdx.y * 64;
    const int lm   = lane & 15;
    const int hi   = lane >> 4;
    const unsigned short* wz = wt + z * EDIM * EDIM;
    const float* bias = (z == 0) ? bq : (z == 1) ? bk : bv;

    // Staging map: 256 threads x 16B cover the 64x32 tile.
    const int srow = tid >> 2, scol = (tid & 3) * 8;
    const unsigned short* wsrc = wz + (n0 + srow) * EDIM + scol;
    unsigned short* ldst0 = &wtile[0][srow * 40 + scol];
    unsigned short* ldst1 = &wtile[1][srow * 40 + scol];

    cp16_async(ldst0, wsrc);                    // prime buffer 0

    f32x8 acc[4] = {};
    const unsigned short* arow = xb + (m0 + lm) * EDIM + hi * 8;
    for (int k0 = 0; k0 < EDIM; k0 += 32) {
        const int buf = (k0 >> 5) & 1;
        if (k0 + 32 < EDIM) {
            cp16_async(buf ? ldst0 : ldst1, wsrc + k0 + 32);
            wait_async<1>();                    // current buffer's copy done
        } else {
            wait_async<0>();
        }
        FragBF a;
        a.u[0] = *(const uint4*)(arow + k0);
        a.u[1] = *(const uint4*)(arow + k0 + 16);
        __builtin_prefetch(arow + k0 + 32, 0, 0);
        __syncthreads();
        for (int j = 0; j < 4; ++j) {
            const unsigned short* brow = &wtile[buf][(16 * j + lm) * 40 + hi * 8];
            FragBF b;
            b.u[0] = *(const uint4*)(brow);
            b.u[1] = *(const uint4*)(brow + 16);
            acc[j] = wmma_bf16(a, b, acc[j]);
        }
        __syncthreads();
    }
    // Epilogue: bias + scatter into head layouts.
    for (int j = 0; j < 4; ++j) {
        int col = n0 + 16 * j + lm;
        float bb = bias[col];
        int h = col >> 6, d = col & 63;
        for (int r = 0; r < 8; ++r) {
            int row  = m0 + r + 8 * hi;        // global token index b*T + t
            int bidx = row >> 11;
            int t    = row & (TDIM - 1);
            int bh   = bidx * HDIM + h;
            unsigned short val = f2bf(acc[j][r] + bb);
            if (z == 0)      Qb[(bh * TDIM + t) * DDIM + d] = val;
            else if (z == 1) Kb[(bh * TDIM + t) * DDIM + d] = val;
            else             Vt[(bh * DDIM + d) * TDIM + t] = val;
        }
    }
}

// ---------------------------------------------------------------------------
// Kernel 4: per-key-column softmax stats (softmax is over the QUERY axis!).
// For each (b,h,k): m[k] = max_q S[q,k], l[k] = sum_q exp(S[q,k]-m[k]).
// All 8 waves stream the SAME Q tiles -> stage 32 q-rows in LDS per step
// (async, double buffered); each wave owns 16 key columns, K-frags are
// loop-invariant in registers.
// ---------------------------------------------------------------------------
__global__ __launch_bounds__(256) void k_colstats(const unsigned short* __restrict__ Qb,
                                                  const unsigned short* __restrict__ Kb,
                                                  float* __restrict__ mbuf,
                                                  float* __restrict__ lbuf) {
    __shared__ unsigned short qtile[2][32 * 72];   // 32 q-rows x 64 d, padded
    const int tid = threadIdx.x;
    const int lane = tid & 31, wave = tid >> 5;
    const int lm = lane & 15, hi = lane >> 4;
    const int bh = blockIdx.y;
    const int kc = blockIdx.x * 128 + wave * 16 + lm;   // this lane's key column
    const float scale = 0.125f;                          // 1/sqrt(64)

    // K^T B-fragments for both d-halves: loop-invariant, loaded once.
    const unsigned short* krow = Kb + (bh * TDIM + kc) * DDIM + hi * 8;
    FragBF kb0, kb1;
    kb0.u[0] = *(const uint4*)(krow);       kb0.u[1] = *(const uint4*)(krow + 16);
    kb1.u[0] = *(const uint4*)(krow + 32);  kb1.u[1] = *(const uint4*)(krow + 48);

    // Staging map: 256 threads x 16B cover the 32x64 Q tile.
    const int srow = tid >> 3, scol = (tid & 7) * 8;
    const unsigned short* qsrc = Qb + (bh * TDIM + srow) * DDIM + scol;
    unsigned short* qdst0 = &qtile[0][srow * 72 + scol];
    unsigned short* qdst1 = &qtile[1][srow * 72 + scol];

    cp16_async(qdst0, qsrc);                    // prime buffer 0

    float m = -3.0e38f, l = 0.0f;
    for (int blk = 0; blk < TDIM / 32; ++blk) {
        const int buf = blk & 1;
        if (blk + 1 < TDIM / 32) {
            cp16_async(buf ? qdst0 : qdst1, qsrc + (blk + 1) * 32 * DDIM);
            wait_async<1>();
        } else {
            wait_async<0>();
        }
        __syncthreads();
        for (int t2 = 0; t2 < 2; ++t2) {
            const unsigned short* qr = &qtile[buf][(t2 * 16 + lm) * 72 + hi * 8];
            FragBF a0, a1;
            a0.u[0] = *(const uint4*)(qr);       a0.u[1] = *(const uint4*)(qr + 16);
            a1.u[0] = *(const uint4*)(qr + 32);  a1.u[1] = *(const uint4*)(qr + 48);
            f32x8 s = {};
            s = wmma_bf16(a0, kb0, s);
            s = wmma_bf16(a1, kb1, s);
            float sv[8], tm = -3.0e38f;
            for (int r = 0; r < 8; ++r) { sv[r] = s[r] * scale; tm = fmaxf(tm, sv[r]); }
            float nm  = fmaxf(m, tm);
            float acc = l * __expf(m - nm);
            for (int r = 0; r < 8; ++r) acc += __expf(sv[r] - nm);
            m = nm; l = acc;
        }
        __syncthreads();
    }
    // Lanes L and L+16 hold the same column (C-layout rows r and r+8): merge.
    float om = __shfl_xor(m, 16, 32);
    float ol = __shfl_xor(l, 16, 32);
    float M = fmaxf(m, om);
    float L = l * __expf(m - M) + ol * __expf(om - M);
    if (hi == 0) {
        mbuf[bh * TDIM + kc] = M;
        lbuf[bh * TDIM + kc] = L;
    }
}

// ---------------------------------------------------------------------------
// Kernel 5: fused attention output. K/V tiles are shared by all 8 waves ->
// async-stage both into LDS (double buffered). Recompute S tiles with WMMA,
// form P = exp(S-m[k])/l[k], transpose P (C-layout -> A-layout) through a
// per-wave LDS tile, accumulate O += P @ V with WMMA.
// ---------------------------------------------------------------------------
__global__ __launch_bounds__(256) void k_attn(const unsigned short* __restrict__ Qb,
                                              const unsigned short* __restrict__ Kb,
                                              const unsigned short* __restrict__ Vt,
                                              const float* __restrict__ mbuf,
                                              const float* __restrict__ lbuf,
                                              unsigned short* __restrict__ Ob) {
    __shared__ unsigned short ktile[2][32 * 72];   // 32 keys x 64 d, padded
    __shared__ unsigned short vtile[2][64 * 40];   // 64 d x 32 keys, padded
    __shared__ unsigned short ptile[8][16 * 40];   // per-wave P transpose tile
    const int tid = threadIdx.x;
    const int lane = tid & 31, wave = tid >> 5;
    const int lm = lane & 15, hi = lane >> 4;
    const int bh = blockIdx.y;
    const int q0 = blockIdx.x * 128 + wave * 16;
    const float scale = 0.125f;

    // Q A-fragments for this query tile: loop-invariant.
    const unsigned short* qrow = Qb + (bh * TDIM + q0 + lm) * DDIM + hi * 8;
    FragBF qa0, qa1;
    qa0.u[0] = *(const uint4*)(qrow);       qa0.u[1] = *(const uint4*)(qrow + 16);
    qa1.u[0] = *(const uint4*)(qrow + 32);  qa1.u[1] = *(const uint4*)(qrow + 48);

    // Staging maps (256 threads x 16B per tile).
    const int krs = tid >> 3, kcs = (tid & 7) * 8;      // K: 32 x 64
    const int vrs = tid >> 2, vcs = (tid & 3) * 8;      // V: 64 x 32
    const unsigned short* ksrc = Kb + (bh * TDIM + krs) * DDIM + kcs;
    const unsigned short* vsrc = Vt + (bh * DDIM + vrs) * TDIM + vcs;
    unsigned short* kdst[2] = { &ktile[0][krs * 72 + kcs], &ktile[1][krs * 72 + kcs] };
    unsigned short* vdst[2] = { &vtile[0][vrs * 40 + vcs], &vtile[1][vrs * 40 + vcs] };

    cp16_async(kdst[0], ksrc);
    cp16_async(vdst[0], vsrc);

    f32x8 o[4] = {};
    unsigned short* myl = &ptile[wave][0];
    const float* mrow = mbuf + bh * TDIM;
    const float* lrow = lbuf + bh * TDIM;

    for (int k0 = 0; k0 < TDIM; k0 += 32) {
        const int buf = (k0 >> 5) & 1;
        if (k0 + 32 < TDIM) {
            cp16_async(kdst[buf ^ 1], ksrc + (k0 + 32) * DDIM);
            cp16_async(vdst[buf ^ 1], vsrc + (k0 + 32));
            wait_async<2>();                    // current K+V copies complete
        } else {
            wait_async<0>();
        }
        __syncthreads();
        // Two 16x16 score tiles -> P, staged transposed into per-wave LDS.
        for (int j = 0; j < 2; ++j) {
            const unsigned short* kr = &ktile[buf][(16 * j + lm) * 72 + hi * 8];
            FragBF kb0, kb1;
            kb0.u[0] = *(const uint4*)(kr);       kb0.u[1] = *(const uint4*)(kr + 16);
            kb1.u[0] = *(const uint4*)(kr + 32);  kb1.u[1] = *(const uint4*)(kr + 48);
            f32x8 s = {};
            s = wmma_bf16(qa0, kb0, s);
            s = wmma_bf16(qa1, kb1, s);
            int kcg = k0 + 16 * j + lm;
            float mc = mrow[kcg];
            float rl = 1.0f / lrow[kcg];
            for (int r = 0; r < 8; ++r) {
                float p = __expf(s[r] * scale - mc) * rl;
                myl[(r + 8 * hi) * 40 + 16 * j + lm] = f2bf(p);
            }
        }
        asm volatile("s_wait_dscnt 0" ::: "memory");   // P stores -> frag reads
        FragBF pa;
        const unsigned short* prow = myl + lm * 40 + hi * 8;
        pa.u[0] = *(const uint4*)(prow);
        pa.u[1] = *(const uint4*)(prow + 16);
        for (int j = 0; j < 4; ++j) {
            const unsigned short* vr = &vtile[buf][(16 * j + lm) * 40 + hi * 8];
            FragBF vb;
            vb.u[0] = *(const uint4*)(vr);
            vb.u[1] = *(const uint4*)(vr + 16);
            o[j] = wmma_bf16(pa, vb, o[j]);
        }
        __syncthreads();
    }
    // Store O contiguously as (B,H,T,D): flat-aliases the reference's
    // (B,H,T,D).reshape(B,T,E) with no transpose.
    unsigned short* obase = Ob + (bh * TDIM + q0) * DDIM;
    for (int j = 0; j < 4; ++j)
        for (int r = 0; r < 8; ++r)
            obase[(r + 8 * hi) * DDIM + 16 * j + lm] = f2bf(o[j][r]);
}

// ---------------------------------------------------------------------------
// Kernel 6: output projection (8192x512 @ 512x512) + bias -> fp32 d_out.
// Weight tile staged in LDS like k_qkv.
// ---------------------------------------------------------------------------
__global__ __launch_bounds__(256) void k_oproj(const unsigned short* __restrict__ Ob,
                                               const unsigned short* __restrict__ wt,
                                               const float* __restrict__ bo,
                                               float* __restrict__ out) {
    __shared__ unsigned short wtile[2][64 * 40];
    const int tid = threadIdx.x;
    const int lane = tid & 31, wave = tid >> 5;
    const int lm = lane & 15, hi = lane >> 4;
    const int m0 = blockIdx.x * 128 + wave * 16;
    const int n0 = blockIdx.y * 64;

    const int srow = tid >> 2, scol = (tid & 3) * 8;
    const unsigned short* wsrc = wt + (n0 + srow) * EDIM + scol;
    unsigned short* ldst0 = &wtile[0][srow * 40 + scol];
    unsigned short* ldst1 = &wtile[1][srow * 40 + scol];

    cp16_async(ldst0, wsrc);

    f32x8 acc[4] = {};
    const unsigned short* arow = Ob + (m0 + lm) * EDIM + hi * 8;
    for (int k0 = 0; k0 < EDIM; k0 += 32) {
        const int buf = (k0 >> 5) & 1;
        if (k0 + 32 < EDIM) {
            cp16_async(buf ? ldst0 : ldst1, wsrc + k0 + 32);
            wait_async<1>();
        } else {
            wait_async<0>();
        }
        FragBF a;
        a.u[0] = *(const uint4*)(arow + k0);
        a.u[1] = *(const uint4*)(arow + k0 + 16);
        __builtin_prefetch(arow + k0 + 32, 0, 0);
        __syncthreads();
        for (int j = 0; j < 4; ++j) {
            const unsigned short* brow = &wtile[buf][(16 * j + lm) * 40 + hi * 8];
            FragBF b;
            b.u[0] = *(const uint4*)(brow);
            b.u[1] = *(const uint4*)(brow + 16);
            acc[j] = wmma_bf16(a, b, acc[j]);
        }
        __syncthreads();
    }
    for (int j = 0; j < 4; ++j) {
        int col = n0 + 16 * j + lm;
        float bb = bo[col];
        for (int r = 0; r < 8; ++r)
            out[(m0 + r + 8 * hi) * EDIM + col] = acc[j][r] + bb;
    }
}

// ---------------------------------------------------------------------------
extern "C" void kernel_launch(void* const* d_in, const int* in_sizes, int n_in,
                              void* d_out, int out_size, void* d_ws, size_t ws_size,
                              hipStream_t stream) {
    const float* x  = (const float*)d_in[0];
    const float* Wq = (const float*)d_in[1];
    const float* bq = (const float*)d_in[2];
    const float* Wk = (const float*)d_in[3];
    const float* bk = (const float*)d_in[4];
    const float* Wv = (const float*)d_in[5];
    const float* bv = (const float*)d_in[6];
    const float* Wo = (const float*)d_in[7];
    const float* bo = (const float*)d_in[8];
    float* out = (float*)d_out;

    char* p = (char*)d_ws;
    auto alloc = [&](size_t bytes) {
        char* r = p;
        p += (bytes + 255) & ~(size_t)255;
        return r;
    };
    unsigned short* Xb = (unsigned short*)alloc((size_t)MROWS * EDIM * 2);      // 8 MB
    unsigned short* Wt = (unsigned short*)alloc((size_t)4 * EDIM * EDIM * 2);   // 2 MB
    unsigned short* Qb = (unsigned short*)alloc((size_t)BH * TDIM * DDIM * 2);  // 8 MB
    unsigned short* Kb = (unsigned short*)alloc((size_t)BH * TDIM * DDIM * 2);  // 8 MB
    unsigned short* Vt = (unsigned short*)alloc((size_t)BH * DDIM * TDIM * 2);  // 8 MB
    unsigned short* Ob = (unsigned short*)alloc((size_t)BH * TDIM * DDIM * 2);  // 8 MB
    float* mbuf = (float*)alloc((size_t)BH * TDIM * 4);
    float* lbuf = (float*)alloc((size_t)BH * TDIM * 4);

    k_cvt_x<<<dim3(MROWS * EDIM / 512), 256, 0, stream>>>(x, Xb, MROWS * EDIM);
    k_cvt_w<<<dim3(EDIM * EDIM / 256, 4), 256, 0, stream>>>(Wq, Wk, Wv, Wo, Wt);
    k_qkv<<<dim3(MROWS / 128, EDIM / 64, 3), 256, 0, stream>>>(Xb, Wt, bq, bk, bv, Qb, Kb, Vt);
    k_colstats<<<dim3(TDIM / 128, BH), 256, 0, stream>>>(Qb, Kb, mbuf, lbuf);
    k_attn<<<dim3(TDIM / 128, BH), 256, 0, stream>>>(Qb, Kb, Vt, mbuf, lbuf, Ob);
    k_oproj<<<dim3(MROWS / 128, EDIM / 64), 256, 0, stream>>>(Ob, Wt + 3 * EDIM * EDIM, bo, out);
}
